// Qwen3MoeSparseMoeBlock_33191507263951
// MI455X (gfx1250) — matrix operand order repaired and checked by reading, not compiled, not run on hardware
//
#include <hip/hip_runtime.h>

#define TOK   1024
#define HDIM  2048
#define NEXP  64
#define TOPK  8
#define IDIM  768
#define I2    1536
#define KC    64

typedef __attribute__((ext_vector_type(16))) __bf16 v16bf;
typedef __attribute__((ext_vector_type(8)))  __bf16 v8bf;
typedef __attribute__((ext_vector_type(8)))  float  v8f;
typedef __attribute__((ext_vector_type(4)))  int    v4i;

#if defined(__has_builtin)
#if __has_builtin(__builtin_amdgcn_global_load_async_to_lds_b128)
#define HAS_ASYNC_LDS 1
#endif
#endif
#ifndef HAS_ASYNC_LDS
#define HAS_ASYNC_LDS 0
#endif

#if HAS_ASYNC_LDS
typedef __attribute__((address_space(1))) v4i as1_v4i;
typedef __attribute__((address_space(3))) v4i as3_v4i;
__device__ inline void wait_async0() {
#if __has_builtin(__builtin_amdgcn_s_wait_asynccnt)
  __builtin_amdgcn_s_wait_asynccnt(0);
#else
  asm volatile("s_wait_asynccnt 0x0" ::: "memory");
#endif
}
// copy 16 bf16 (32B) global -> LDS, no VGPR data movement, ASYNCcnt-tracked
__device__ inline void async_cp32(__bf16* ldst, const __bf16* gsrc) {
  __builtin_amdgcn_global_load_async_to_lds_b128((as1_v4i*)gsrc,
                                                 (as3_v4i*)ldst, 0, 0);
  __builtin_amdgcn_global_load_async_to_lds_b128((as1_v4i*)(gsrc + 8),
                                                 (as3_v4i*)(ldst + 8), 0, 0);
}
#endif

__device__ inline v16bf make_frag(const __bf16* p0, const __bf16* p1) {
  v8bf a = *(const v8bf*)p0;
  v8bf b = *(const v8bf*)p1;
  v16bf r;
#pragma unroll
  for (int i = 0; i < 8; ++i) { r[i] = a[i]; r[i + 8] = b[i]; }
  return r;
}

__device__ inline v8f wmma_bf16(v16bf a, v16bf b, v8f c) {
  return __builtin_amdgcn_wmma_f32_16x16x32_bf16(false, a, false, b, (short)0, c,
                                                 false, false);
}

// ---------------- zero out + counters + zero-row ----------------
__global__ void zero_kernel(float* out, int* cnt, __bf16* zrow) {
  size_t i = (size_t)blockIdx.x * blockDim.x + threadIdx.x;
  if (i < (size_t)TOK * HDIM) out[i] = 0.f;
  if (blockIdx.x == 0 && threadIdx.x < NEXP) cnt[threadIdx.x] = 0;
  if (blockIdx.x == 1) {           // 1024 bf16 zero row = 512 dwords
    unsigned int* z = (unsigned int*)zrow;
    z[threadIdx.x] = 0u;
    z[threadIdx.x + 256] = 0u;
  }
}

// ---------------- router: logits, top-8, renorm; fused x -> bf16 ----------------
__global__ void __launch_bounds__(256)
router_kernel(const float* __restrict__ x, const float* __restrict__ wg,
              int* __restrict__ topk_idx, float* __restrict__ topk_w,
              __bf16* __restrict__ xb) {
  __shared__ float xs[HDIM];
  __shared__ float part[4][NEXP];
  __shared__ float lg[NEXP];
  const int t = blockIdx.x, tid = threadIdx.x;
  for (int h = tid; h < HDIM; h += 256) {
    float v = x[(size_t)t * HDIM + h];
    xs[h] = v;
    xb[(size_t)t * HDIM + h] = (__bf16)v;
  }
  __syncthreads();
  const int e = tid & 63, g = tid >> 6;
  float s = 0.f;
  for (int h = g * 512; h < g * 512 + 512; ++h) s += xs[h] * wg[(size_t)h * NEXP + e];
  part[g][e] = s;
  __syncthreads();
  if (tid < NEXP) lg[tid] = part[0][tid] + part[1][tid] + part[2][tid] + part[3][tid];
  __syncthreads();
  if (tid == 0) {
    float sel_w[TOPK]; int sel_i[TOPK]; float lmax = -1e30f;
    for (int k = 0; k < TOPK; ++k) {
      float best = -1e30f; int bi = 0;
      for (int j = 0; j < NEXP; ++j) if (lg[j] > best) { best = lg[j]; bi = j; }
      if (k == 0) lmax = best;
      sel_w[k] = __expf(best - lmax); sel_i[k] = bi;
      lg[bi] = -1e30f;
    }
    float ws = 0.f;
    for (int k = 0; k < TOPK; ++k) ws += sel_w[k];
    float inv = 1.f / ws;
    for (int k = 0; k < TOPK; ++k) {
      topk_idx[t * TOPK + k] = sel_i[k];
      topk_w[t * TOPK + k]   = sel_w[k] * inv;
    }
  }
}

// ---------------- scatter tokens to experts ----------------
__global__ void scatter_kernel(const int* __restrict__ topk_idx,
                               const float* __restrict__ topk_w,
                               int* cnt, int* tok, float* wts) {
  int i = blockIdx.x * blockDim.x + threadIdx.x;
  if (i >= TOK * TOPK) return;
  int e = topk_idx[i];
  int pos = atomicAdd(&cnt[e], 1);
  tok[e * TOK + pos] = i / TOPK;
  wts[e * TOK + pos] = topk_w[i];
}

__global__ void offs_kernel(const int* __restrict__ cnt, int* __restrict__ offs) {
  if (threadIdx.x == 0) {
    int a = 0;
    for (int e = 0; e < NEXP; ++e) { offs[e] = a; a += cnt[e]; }
  }
}

// pack one 8k x 4col f32 subtile into col-major bf16 LDS (four ds_store_b128)
__device__ inline void store_b_subtile(__bf16* lB, const float4* B,
                                       int cg4, int kg) {
  v8bf t0, t1, t2, t3;
#pragma unroll
  for (int r = 0; r < 8; ++r) {
    t0[r] = (__bf16)B[r].x;
    t1[r] = (__bf16)B[r].y;
    t2[r] = (__bf16)B[r].z;
    t3[r] = (__bf16)B[r].w;
  }
  *(v8bf*)&lB[(cg4 + 0) * KC + kg * 8] = t0;
  *(v8bf*)&lB[(cg4 + 1) * KC + kg * 8] = t1;
  *(v8bf*)&lB[(cg4 + 2) * KC + kg * 8] = t2;
  *(v8bf*)&lB[(cg4 + 3) * KC + kg * 8] = t3;
}

// ---------------- GEMM1 + SiLU*up -> act (bf16) ----------------
// block = (expert, m-tile of 128 tokens); 8 waves = 8 m-frags x full 128 B cols
__global__ void __launch_bounds__(256)
gemm1_silu_kernel(const __bf16* __restrict__ xb, const float* __restrict__ w13,
                  const int* __restrict__ tok, const int* __restrict__ cnt,
                  const int* __restrict__ offs, __bf16* __restrict__ act) {
  __shared__ __attribute__((aligned(16))) __bf16 lA[2][128 * KC];
  __shared__ __attribute__((aligned(16))) __bf16 lB[128 * KC];
  __shared__ int lTok[128];
  const int e = blockIdx.x >> 3, tile = blockIdx.x & 7;
  const int n = cnt[e], m0 = tile * 128;
  if (m0 >= n) return;
  const int Mv = min(128, n - m0);
  const int tid = threadIdx.x;
  if (tid < 128) lTok[tid] = (tid < Mv) ? tok[e * TOK + m0 + tid] : 0;
  __syncthreads();
  const float* w13e = w13 + (size_t)e * HDIM * I2;
  const int base = offs[e] + m0;
  const int lane = tid & 31, mf = tid >> 5;
  const int half = lane >> 4, nlo = lane & 15;

  const int r0 = tid >> 2, ac = (tid & 3) * 16;      // A: rows r0, r0+64
  const int kg = tid >> 5, cg4 = (tid & 31) * 4;     // B: 8k-group, 4col-group
  const __bf16* arow0 = xb + (size_t)lTok[r0] * HDIM + ac;
  const __bf16* arow1 = xb + (size_t)lTok[r0 + 64] * HDIM + ac;

  for (int p = 0; p < 12; ++p) {                     // 12 passes x 64 act cols
    const size_t colbase = (cg4 < 64) ? (size_t)(p * 64 + cg4)
                                      : (size_t)(IDIM + p * 64 + (cg4 - 64));
    const float* bcol = w13e + colbase;
    v8f aG[4] = {}, aU[4] = {};
    float4 B[8];
#pragma unroll
    for (int r = 0; r < 8; ++r)
      B[r] = *(const float4*)(bcol + (size_t)(kg * 8 + r) * I2);
#if HAS_ASYNC_LDS
    int cur = 0;
    async_cp32(&lA[0][r0 * KC + ac], arow0);
    async_cp32(&lA[0][(r0 + 64) * KC + ac], arow1);
#else
    uint4 A00 = *(const uint4*)(arow0), A01 = *(const uint4*)(arow0 + 8);
    uint4 A10 = *(const uint4*)(arow1), A11 = *(const uint4*)(arow1 + 8);
#endif
    for (int k0 = 0; k0 < HDIM; k0 += KC) {
      const __bf16* Abuf;
#if HAS_ASYNC_LDS
      wait_async0();
      __syncthreads();
      store_b_subtile(lB, B, cg4, kg);
      if (k0 + KC < HDIM) {
        async_cp32(&lA[cur ^ 1][r0 * KC + ac], arow0 + k0 + KC);
        async_cp32(&lA[cur ^ 1][(r0 + 64) * KC + ac], arow1 + k0 + KC);
#pragma unroll
        for (int r = 0; r < 8; ++r)
          B[r] = *(const float4*)(bcol + (size_t)(k0 + KC + kg * 8 + r) * I2);
      }
      __syncthreads();
      Abuf = (const __bf16*)lA[cur];
      cur ^= 1;
#else
      __syncthreads();
      *(uint4*)&lA[0][r0 * KC + ac]            = A00;
      *(uint4*)&lA[0][r0 * KC + ac + 8]        = A01;
      *(uint4*)&lA[0][(r0 + 64) * KC + ac]     = A10;
      *(uint4*)&lA[0][(r0 + 64) * KC + ac + 8] = A11;
      store_b_subtile(lB, B, cg4, kg);
      __syncthreads();
      if (k0 + KC < HDIM) {
        A00 = *(const uint4*)(arow0 + k0 + KC);
        A01 = *(const uint4*)(arow0 + k0 + KC + 8);
        A10 = *(const uint4*)(arow1 + k0 + KC);
        A11 = *(const uint4*)(arow1 + k0 + KC + 8);
#pragma unroll
        for (int r = 0; r < 8; ++r)
          B[r] = *(const float4*)(bcol + (size_t)(k0 + KC + kg * 8 + r) * I2);
      }
      Abuf = (const __bf16*)lA[0];
#endif
#pragma unroll
      for (int kk = 0; kk < KC; kk += 32) {
        const __bf16* arp = &Abuf[(mf * 16 + nlo) * KC + kk];
        v16bf a = make_frag(arp + half * 16, arp + half * 16 + 8);
#pragma unroll
        for (int f = 0; f < 4; ++f) {
          const __bf16* qg = &lB[(f * 16 + nlo) * KC + kk + half * 16];
          v16bf bg = make_frag(qg, qg + 8);
          aG[f] = wmma_bf16(a, bg, aG[f]);
          const __bf16* qu = &lB[(64 + f * 16 + nlo) * KC + kk + half * 16];
          v16bf bu = make_frag(qu, qu + 8);
          aU[f] = wmma_bf16(a, bu, aU[f]);
        }
      }
    }
    // epilogue: act = silu(gate) * up, store bf16
#pragma unroll
    for (int f = 0; f < 4; ++f)
#pragma unroll
      for (int i = 0; i < 8; ++i) {
        int mloc = mf * 16 + i + 8 * half;
        if (mloc < Mv) {
          int colb = p * 64 + f * 16 + nlo;
          float g0 = aG[f][i], u0 = aU[f][i];
          act[(size_t)(base + mloc) * IDIM + colb] =
              (__bf16)(g0 / (1.f + __expf(-g0)) * u0);
        }
      }
  }
}

// ---------------- GEMM2 + weighted combine into out ----------------
__global__ void __launch_bounds__(256)
gemm2_combine_kernel(const __bf16* __restrict__ act, const float* __restrict__ w2,
                     const int* __restrict__ tok, const float* __restrict__ wts,
                     const int* __restrict__ cnt, const int* __restrict__ offs,
                     const __bf16* __restrict__ zrow, float* __restrict__ out) {
  __shared__ __attribute__((aligned(16))) __bf16 lA[2][128 * KC];
  __shared__ __attribute__((aligned(16))) __bf16 lB[128 * KC];
  __shared__ int lTok[128];
  __shared__ float lW[128];
  const int e = blockIdx.x >> 3, tile = blockIdx.x & 7;
  const int n = cnt[e], m0 = tile * 128;
  if (m0 >= n) return;
  const int Mv = min(128, n - m0);
  const int tid = threadIdx.x;
  if (tid < 128) {
    lTok[tid] = (tid < Mv) ? tok[e * TOK + m0 + tid] : 0;
    lW[tid]   = (tid < Mv) ? wts[e * TOK + m0 + tid] : 0.f;
  }
  __syncthreads();
  const float* w2e = w2 + (size_t)e * IDIM * HDIM;
  const int base = offs[e] + m0;
  const int lane = tid & 31, mf = tid >> 5;
  const int half = lane >> 4, nlo = lane & 15;

  const int r0 = tid >> 2, ac = (tid & 3) * 16;
  const int kg = tid >> 5, cg4 = (tid & 31) * 4;
  const __bf16* arow0 =
      (r0 < Mv) ? act + (size_t)(base + r0) * IDIM + ac : zrow + ac;
  const __bf16* arow1 =
      (r0 + 64 < Mv) ? act + (size_t)(base + r0 + 64) * IDIM + ac : zrow + ac;

  for (int p = 0; p < 16; ++p) {                     // 16 passes x 128 out cols
    const int n0 = p * 128;
    const float* bcol = w2e + n0 + cg4;
    v8f acc[8] = {};
    float4 B[8];
#pragma unroll
    for (int r = 0; r < 8; ++r)
      B[r] = *(const float4*)(bcol + (size_t)(kg * 8 + r) * HDIM);
#if HAS_ASYNC_LDS
    int cur = 0;
    async_cp32(&lA[0][r0 * KC + ac], arow0);
    async_cp32(&lA[0][(r0 + 64) * KC + ac], arow1);
#else
    uint4 A00 = *(const uint4*)(arow0), A01 = *(const uint4*)(arow0 + 8);
    uint4 A10 = *(const uint4*)(arow1), A11 = *(const uint4*)(arow1 + 8);
#endif
    for (int k0 = 0; k0 < IDIM; k0 += KC) {
      const __bf16* Abuf;
#if HAS_ASYNC_LDS
      wait_async0();
      __syncthreads();
      store_b_subtile(lB, B, cg4, kg);
      if (k0 + KC < IDIM) {
        async_cp32(&lA[cur ^ 1][r0 * KC + ac], arow0 + k0 + KC);
        async_cp32(&lA[cur ^ 1][(r0 + 64) * KC + ac], arow1 + k0 + KC);
#pragma unroll
        for (int r = 0; r < 8; ++r)
          B[r] = *(const float4*)(bcol + (size_t)(k0 + KC + kg * 8 + r) * HDIM);
      }
      __syncthreads();
      Abuf = (const __bf16*)lA[cur];
      cur ^= 1;
#else
      __syncthreads();
      *(uint4*)&lA[0][r0 * KC + ac]            = A00;
      *(uint4*)&lA[0][r0 * KC + ac + 8]        = A01;
      *(uint4*)&lA[0][(r0 + 64) * KC + ac]     = A10;
      *(uint4*)&lA[0][(r0 + 64) * KC + ac + 8] = A11;
      store_b_subtile(lB, B, cg4, kg);
      __syncthreads();
      if (k0 + KC < IDIM) {
        A00 = *(const uint4*)(arow0 + k0 + KC);
        A01 = *(const uint4*)(arow0 + k0 + KC + 8);
        A10 = *(const uint4*)(arow1 + k0 + KC);
        A11 = *(const uint4*)(arow1 + k0 + KC + 8);
#pragma unroll
        for (int r = 0; r < 8; ++r)
          B[r] = *(const float4*)(bcol + (size_t)(k0 + KC + kg * 8 + r) * HDIM);
      }
      Abuf = (const __bf16*)lA[0];
#endif
#pragma unroll
      for (int kk = 0; kk < KC; kk += 32) {
        const __bf16* arp = &Abuf[(mf * 16 + nlo) * KC + kk];
        v16bf a = make_frag(arp + half * 16, arp + half * 16 + 8);
#pragma unroll
        for (int f = 0; f < 8; ++f) {
          const __bf16* q = &lB[(f * 16 + nlo) * KC + kk + half * 16];
          v16bf b = make_frag(q, q + 8);
          acc[f] = wmma_bf16(a, b, acc[f]);
        }
      }
    }
#pragma unroll
    for (int i = 0; i < 8; ++i) {
      int mloc = mf * 16 + i + 8 * half;
      if (mloc < Mv) {
        int t = lTok[mloc]; float wgt = lW[mloc];
#pragma unroll
        for (int f = 0; f < 8; ++f)
          atomicAdd(&out[(size_t)t * HDIM + n0 + f * 16 + nlo], acc[f][i] * wgt);
      }
    }
  }
}

extern "C" void kernel_launch(void* const* d_in, const int* in_sizes, int n_in,
                              void* d_out, int out_size, void* d_ws, size_t ws_size,
                              hipStream_t stream) {
  const float* x   = (const float*)d_in[0];   // (1,1024,2048)
  const float* wg  = (const float*)d_in[1];   // (2048,64)
  const float* w13 = (const float*)d_in[2];   // (64,2048,1536)
  const float* w2  = (const float*)d_in[3];   // (64,768,2048)
  float* out = (float*)d_out;

  char* p = (char*)d_ws;
  auto carve = [&](size_t bytes) -> char* {
    char* r = p; p += (bytes + 255) & ~(size_t)255; return r;
  };
  int*    topk_idx = (int*)   carve((size_t)TOK * TOPK * 4);
  float*  topk_w   = (float*) carve((size_t)TOK * TOPK * 4);
  int*    cnt      = (int*)   carve(NEXP * 4);
  int*    offs     = (int*)   carve(NEXP * 4);
  int*    tokl     = (int*)   carve((size_t)NEXP * TOK * 4);
  float*  wtsl     = (float*) carve((size_t)NEXP * TOK * 4);
  __bf16* xb       = (__bf16*)carve((size_t)TOK * HDIM * 2);
  __bf16* actb     = (__bf16*)carve((size_t)TOK * TOPK * IDIM * 2);
  __bf16* zrow     = (__bf16*)carve((size_t)1024 * 2);

  zero_kernel<<<(TOK * HDIM + 255) / 256, 256, 0, stream>>>(out, cnt, zrow);
  router_kernel<<<TOK, 256, 0, stream>>>(x, wg, topk_idx, topk_w, xb);
  scatter_kernel<<<(TOK * TOPK + 255) / 256, 256, 0, stream>>>(topk_idx, topk_w,
                                                               cnt, tokl, wtsl);
  offs_kernel<<<1, 64, 0, stream>>>(cnt, offs);
  gemm1_silu_kernel<<<NEXP * 8, 256, 0, stream>>>(xb, w13, tokl, cnt, offs, actb);
  gemm2_combine_kernel<<<NEXP * 8, 256, 0, stream>>>(actb, w2, tokl, wtsl, cnt,
                                                     offs, zrow, out);
}
